// Prototype1_64072322122268
// MI455X (gfx1250) — compile-verified, matrix operand order checked by reading, not run on hardware
//
#include <hip/hip_runtime.h>

// ---------------------------------------------------------------------------
// MI455X (gfx1250) implementation.
// Heavy GEMMs use v_wmma_f32_16x16x32_bf16 (bf16 operands, fp32 accum).
// Everything numerically sensitive stays fp32 VALU.
// GEMM wave tile: 32x64 (2 A frags x 4 B frags -> 8 WMMAs per 32-wide k-step).
// ---------------------------------------------------------------------------

typedef __attribute__((ext_vector_type(16))) __bf16 v16bf;
typedef __attribute__((ext_vector_type(8)))  __bf16 v8bf;
typedef __attribute__((ext_vector_type(8)))  float  v8f;

// ---------------------------------------------------------------------------
// Fragment loader: 16-bit A/B matrix layout for V_WMMA_*_16X16X32_* (wave32).
// lane 0-15 : row/col = lane, K = [k0..k0+7] (j0..7), [k0+16..k0+23] (j8..15)
// lane 16-31: row/col = lane-16, K = [k0+8..k0+15], [k0+24..k0+31]
// kb = k0 + (lane>=16 ? 8 : 0); both 8-element chunks are 16B aligned.
// ---------------------------------------------------------------------------
__device__ __forceinline__ v16bf frag_load(const __bf16* __restrict__ row, int kb) {
  v8bf lo = *(const v8bf*)(row + kb);
  v8bf hi = *(const v8bf*)(row + kb + 16);
  return __builtin_shufflevector(lo, hi, 0,1,2,3,4,5,6,7,8,9,10,11,12,13,14,15);
}

// ---------------------------------------------------------------------------
// Generic batched bf16 WMMA GEMM: C[M,N] = A[M,K] * Bt[N,K]^T (+bias, relu).
// One wave computes a 32x64 output tile: 2 A fragments x 4 B fragments,
// 8 WMMAs per k-step, 12 B128 loads per 8 WMMAs.
// blockDim = 128 (4 waves). grid.x = (M/32)*(N/64)/4, grid.y = batch.
// ---------------------------------------------------------------------------
__global__ void k_gemm_bf16(const __bf16* __restrict__ A, long lda, long ab,
                            const __bf16* __restrict__ Bt, long ldb, long bb,
                            void* __restrict__ Cp, long ldc, long cb, int c_bf16,
                            const float* __restrict__ bias, long biasb,
                            int M, int N, int K, int relu)
{
  const int batch = blockIdx.y;
  A  += (long)batch * ab;
  Bt += (long)batch * bb;

  const int lane = threadIdx.x & 31;
  const int wid  = blockIdx.x * (blockDim.x >> 5) + (threadIdx.x >> 5);
  const int mtiles = M >> 5;                 // 32-row tiles
  const int mt = wid % mtiles;
  const int ng = wid / mtiles;
  if (ng * 64 >= N) return;

  const int hsel = lane >> 4;   // 0 or 1 (lane half)
  const int l15  = lane & 15;

  const __bf16* arow0 = A + (long)(mt * 32 +  0 + l15) * lda;
  const __bf16* arow1 = A + (long)(mt * 32 + 16 + l15) * lda;
  const __bf16* brow0 = Bt + (long)(ng * 64 +  0 + l15) * ldb;
  const __bf16* brow1 = Bt + (long)(ng * 64 + 16 + l15) * ldb;
  const __bf16* brow2 = Bt + (long)(ng * 64 + 32 + l15) * ldb;
  const __bf16* brow3 = Bt + (long)(ng * 64 + 48 + l15) * ldb;

  v8f c00 = {}; v8f c01 = {}; v8f c02 = {}; v8f c03 = {};
  v8f c10 = {}; v8f c11 = {}; v8f c12 = {}; v8f c13 = {};

  for (int k0 = 0; k0 < K; k0 += 32) {
    const int kb = k0 + hsel * 8;
    v16bf a0 = frag_load(arow0, kb);
    v16bf a1 = frag_load(arow1, kb);
    v16bf b0 = frag_load(brow0, kb);
    v16bf b1 = frag_load(brow1, kb);
    v16bf b2 = frag_load(brow2, kb);
    v16bf b3 = frag_load(brow3, kb);
    c00 = __builtin_amdgcn_wmma_f32_16x16x32_bf16(false, a0, false, b0, (short)0, c00, false, false);
    c01 = __builtin_amdgcn_wmma_f32_16x16x32_bf16(false, a0, false, b1, (short)0, c01, false, false);
    c02 = __builtin_amdgcn_wmma_f32_16x16x32_bf16(false, a0, false, b2, (short)0, c02, false, false);
    c03 = __builtin_amdgcn_wmma_f32_16x16x32_bf16(false, a0, false, b3, (short)0, c03, false, false);
    c10 = __builtin_amdgcn_wmma_f32_16x16x32_bf16(false, a1, false, b0, (short)0, c10, false, false);
    c11 = __builtin_amdgcn_wmma_f32_16x16x32_bf16(false, a1, false, b1, (short)0, c11, false, false);
    c12 = __builtin_amdgcn_wmma_f32_16x16x32_bf16(false, a1, false, b2, (short)0, c12, false, false);
    c13 = __builtin_amdgcn_wmma_f32_16x16x32_bf16(false, a1, false, b3, (short)0, c13, false, false);
  }

  // C/D layout: VGPR i -> (m = tile_m + i + 8*hsel, n = ng*64 + 16*j + l15)
  const float* bv = bias ? (bias + (long)batch * biasb) : nullptr;
  #pragma unroll
  for (int mi = 0; mi < 2; ++mi) {
    const int mbase = mt * 32 + mi * 16 + hsel * 8;
    #pragma unroll
    for (int i = 0; i < 8; ++i) {
      const int m = mbase + i;
      #pragma unroll
      for (int j = 0; j < 4; ++j) {
        const int n = ng * 64 + j * 16 + l15;
        float val;
        if (mi == 0) val = (j == 0 ? c00[i] : j == 1 ? c01[i] : j == 2 ? c02[i] : c03[i]);
        else         val = (j == 0 ? c10[i] : j == 1 ? c11[i] : j == 2 ? c12[i] : c13[i]);
        if (bv)   val += bv[n];
        if (relu) val = val > 0.f ? val : 0.f;
        if (c_bf16) {
          __bf16* C = (__bf16*)Cp + (long)batch * cb;
          C[(long)m * ldc + n] = (__bf16)val;
        } else {
          float* C = (float*)Cp + (long)batch * cb;
          C[(long)m * ldc + n] = val;
        }
      }
    }
  }
}

// ---------------------------------------------------------------------------
// Transpose fp32 [K,N] -> bf16 [N,K] (batched), 32x32 LDS tiles.
// block(32,8), grid(N/32, K/32, batch)
// ---------------------------------------------------------------------------
__global__ void k_transpose_bf16(const float* __restrict__ in, long inb,
                                 __bf16* __restrict__ out, long outb,
                                 int K, int N)
{
  __shared__ float tile[32][33];
  const int batch = blockIdx.z;
  in  += (long)batch * inb;
  out += (long)batch * outb;
  const int n0 = blockIdx.x * 32, k0 = blockIdx.y * 32;
  const int tx = threadIdx.x, ty = threadIdx.y;
  #pragma unroll
  for (int j = 0; j < 32; j += 8)
    tile[ty + j][tx] = in[(long)(k0 + ty + j) * N + n0 + tx];
  __syncthreads();
  #pragma unroll
  for (int j = 0; j < 32; j += 8)
    out[(long)(n0 + ty + j) * K + k0 + tx] = (__bf16)tile[tx][ty + j];
}

// ---------------------------------------------------------------------------
// Conv1: [256,1,128,128] -> conv3x3 SAME + relu + maxpool2 -> [256,16,64,64]
// One thread per pooled output. t = ((b*16+c)*64+py)*64+px.
// ---------------------------------------------------------------------------
__global__ void k_conv1(const float* __restrict__ x, const float* __restrict__ w,
                        const float* __restrict__ bias, float* __restrict__ outp)
{
  const int t  = blockIdx.x * 256 + threadIdx.x;
  const int px = t & 63;
  const int py = (t >> 6) & 63;
  const int c  = (t >> 12) & 15;
  const int b  = t >> 16;

  float wr[9];
  #pragma unroll
  for (int i = 0; i < 9; ++i) wr[i] = w[c * 9 + i];
  const float bc = bias[c];
  const float* img = x + (long)b * 16384;

  float mx = 0.f;  // relu outputs are >= 0
  #pragma unroll
  for (int dy = 0; dy < 2; ++dy) {
    #pragma unroll
    for (int dx = 0; dx < 2; ++dx) {
      const int Y = 2 * py + dy, X = 2 * px + dx;
      float s = bc;
      #pragma unroll
      for (int ky = 0; ky < 3; ++ky) {
        const int iy = Y + ky - 1;
        if (iy < 0 || iy >= 128) continue;
        #pragma unroll
        for (int kx = 0; kx < 3; ++kx) {
          const int ix = X + kx - 1;
          if (ix < 0 || ix >= 128) continue;
          s += wr[ky * 3 + kx] * img[iy * 128 + ix];
        }
      }
      s = s > 0.f ? s : 0.f;
      mx = s > mx ? s : mx;
    }
  }
  outp[t] = mx;
}

// ---------------------------------------------------------------------------
// Conv2: [256,16,64,64] -> conv3x3 SAME (16->32ch) + relu + maxpool2
// -> tokens bf16 written into z[:, 0:1024].
// Block: one (b, oc, y-quarter); input tile 16x18x66 staged in dynamic LDS.
// ---------------------------------------------------------------------------
__global__ void k_conv2(const float* __restrict__ in, const float* __restrict__ w,
                        const float* __restrict__ bias, __bf16* __restrict__ z)
{
  extern __shared__ float smem[];
  float* tile = smem;                 // 16*18*66 = 19008 floats
  __shared__ float wsm[144];

  const int bi = blockIdx.x;
  const int qo = bi & 3;              // y-quarter
  const int oc = (bi >> 2) & 31;
  const int b  = bi >> 7;
  const int tid = threadIdx.x;

  if (tid < 144) wsm[tid] = w[oc * 144 + tid];

  const int y0c = qo * 16 - 1;        // conv-space Y of tile row 0
  for (int i = tid; i < 16 * 18 * 66; i += 256) {
    const int ic = i / (18 * 66);
    const int r  = i - ic * (18 * 66);
    const int yy = r / 66;
    const int xx = r - yy * 66;
    const int Y = y0c + yy, X = xx - 1;
    float v = 0.f;
    if (Y >= 0 && Y < 64 && X >= 0 && X < 64)
      v = in[(((long)b * 16 + ic) * 64 + Y) * 64 + X];
    tile[i] = v;
  }
  __syncthreads();

  const int ly = tid >> 5;            // 0..7 pooled rows in quarter
  const int px = tid & 31;
  const int py = qo * 8 + ly;
  const float bc = bias[oc];

  float mx = 0.f;
  #pragma unroll
  for (int dy = 0; dy < 2; ++dy) {
    #pragma unroll
    for (int dx = 0; dx < 2; ++dx) {
      const int ty0 = 2 * ly + dy;
      const int tx0 = 2 * px + dx;
      float s = bc;
      for (int ic = 0; ic < 16; ++ic) {
        const float* tp = tile + (ic * 18 + ty0) * 66 + tx0;
        const float* wp = wsm + ic * 9;
        #pragma unroll
        for (int ky = 0; ky < 3; ++ky) {
          s += wp[ky * 3 + 0] * tp[ky * 66 + 0];
          s += wp[ky * 3 + 1] * tp[ky * 66 + 1];
          s += wp[ky * 3 + 2] * tp[ky * 66 + 2];
        }
      }
      s = s > 0.f ? s : 0.f;
      mx = s > mx ? s : mx;
    }
  }
  // z row = b*32+oc, col = py*32+px, row stride 2048 (concat buffer)
  z[((long)b * 32 + oc) * 2048 + py * 32 + px] = (__bf16)mx;
}

// ---------------------------------------------------------------------------
// Attention: per (b,h) block. K,V staged in dynamic LDS; fp32 softmax.
// ctx written bf16. grid = 1024, block = 256,
// dyn smem = (8192 + 8192 + 1024) * 4 bytes.
// ---------------------------------------------------------------------------
__global__ void k_attention(const float* __restrict__ q, const float* __restrict__ k,
                            const float* __restrict__ v, __bf16* __restrict__ ctx)
{
  extern __shared__ float smem[];
  float* ksm = smem;            // [32][256]
  float* vsm = smem + 8192;     // [32][256]
  float* scm = smem + 16384;    // [32][32]

  const int b = blockIdx.x >> 2;
  const int h = blockIdx.x & 3;
  const int tid = threadIdx.x;
  const long base = ((long)b * 32) * 1024 + h * 256;   // element offset of (b, e=0, h, d=0)

  for (int i = tid; i < 2048; i += 256) {              // 2048 float4 per matrix
    const int row = i >> 6, c4 = i & 63;
    ((float4*)ksm)[i] = *(const float4*)(k + base + (long)row * 1024 + c4 * 4);
    ((float4*)vsm)[i] = *(const float4*)(v + base + (long)row * 1024 + c4 * 4);
  }
  __syncthreads();

  // scores: thread -> row s, 4 columns
  {
    const int s  = tid >> 3;
    const int t0 = (tid & 7) * 4;
    const float* qr = q + base + (long)s * 1024;
    float a0 = 0.f, a1 = 0.f, a2 = 0.f, a3 = 0.f;
    for (int d = 0; d < 256; ++d) {
      const float qd = qr[d];
      a0 += qd * ksm[(t0 + 0) * 256 + d];
      a1 += qd * ksm[(t0 + 1) * 256 + d];
      a2 += qd * ksm[(t0 + 2) * 256 + d];
      a3 += qd * ksm[(t0 + 3) * 256 + d];
    }
    scm[s * 32 + t0 + 0] = a0 * 0.0625f;   // 1/sqrt(256)
    scm[s * 32 + t0 + 1] = a1 * 0.0625f;
    scm[s * 32 + t0 + 2] = a2 * 0.0625f;
    scm[s * 32 + t0 + 3] = a3 * 0.0625f;
  }
  __syncthreads();

  if (tid < 32) {
    float mx = -3.402823e38f;
    for (int t = 0; t < 32; ++t) mx = scm[tid * 32 + t] > mx ? scm[tid * 32 + t] : mx;
    float sum = 0.f;
    for (int t = 0; t < 32; ++t) {
      const float e = expf(scm[tid * 32 + t] - mx);
      scm[tid * 32 + t] = e;
      sum += e;
    }
    const float inv = 1.f / sum;
    for (int t = 0; t < 32; ++t) scm[tid * 32 + t] *= inv;
  }
  __syncthreads();

  // ctx: thread -> row s, 32 d-columns
  {
    const int s  = tid >> 3;
    const int d0 = (tid & 7) * 32;
    for (int d = d0; d < d0 + 32; ++d) {
      float sum = 0.f;
      #pragma unroll
      for (int t = 0; t < 32; ++t) sum += scm[s * 32 + t] * vsm[t * 256 + d];
      ctx[base + (long)s * 1024 + d] = (__bf16)sum;
    }
  }
}

// ---------------------------------------------------------------------------
// Expert second layer: eo[b, e*64+o] = relu(h1[e,b,:] . W2[e,:,o] + b2[e,o])
// Thread per (b,e,o); o contiguous across threads -> coalesced W2 reads.
// ---------------------------------------------------------------------------
__global__ void k_expert2(const float* __restrict__ h1, const float* __restrict__ W2,
                          const float* __restrict__ b2, float* __restrict__ eo)
{
  const int t = blockIdx.x * 256 + threadIdx.x;   // 524288 total
  const int o = t & 63;
  const int e = (t >> 6) & 31;
  const int b = t >> 11;
  const float* hrow = h1 + ((long)e * 256 + b) * 512;
  const float* wcol = W2 + (long)e * 512 * 64 + o;
  float sum = b2[e * 64 + o];
  for (int hh = 0; hh < 512; ++hh) sum += hrow[hh] * wcol[(long)hh * 64];
  eo[(long)b * 2048 + e * 64 + o] = sum > 0.f ? sum : 0.f;
}

// ---------------------------------------------------------------------------
// Final head: out[b,o] = eo[b,:] . Wf[:,o] + bf[o]. 512 threads.
// ---------------------------------------------------------------------------
__global__ void k_final(const float* __restrict__ eo, const float* __restrict__ Wf,
                        const float* __restrict__ bfv, float* __restrict__ out)
{
  const int t = blockIdx.x * 256 + threadIdx.x;
  if (t >= 512) return;
  const int o = t & 1;
  const int b = t >> 1;
  const float* row = eo + (long)b * 2048;
  float sum = bfv[o];
  for (int j = 0; j < 2048; ++j) sum += row[j] * Wf[(long)j * 2 + o];
  out[(long)b * 2 + o] = sum;
}

// ---------------------------------------------------------------------------

extern "C" void kernel_launch(void* const* d_in, const int* in_sizes, int n_in,
                              void* d_out, int out_size, void* d_ws, size_t ws_size,
                              hipStream_t stream) {
  const float* x    = (const float*)d_in[0];
  const float* c1w  = (const float*)d_in[1];
  const float* c1b  = (const float*)d_in[2];
  const float* c2w  = (const float*)d_in[3];
  const float* c2b  = (const float*)d_in[4];
  const float* wq   = (const float*)d_in[5];
  const float* wk   = (const float*)d_in[6];
  const float* wv   = (const float*)d_in[7];
  const float* wo   = (const float*)d_in[8];
  const float* W1   = (const float*)d_in[9];
  const float* b1   = (const float*)d_in[10];
  const float* W2   = (const float*)d_in[11];
  const float* b2   = (const float*)d_in[12];
  const float* Wf   = (const float*)d_in[13];
  const float* bfv  = (const float*)d_in[14];
  float* out = (float*)d_out;

  // ---- workspace layout (bytes) ----
  char* ws = (char*)d_ws;
  size_t off = 0;
  auto take = [&](size_t bytes) { char* p = ws + off; off += (bytes + 255) & ~(size_t)255; return p; };
  __bf16* z       = (__bf16*)take((size_t)8192 * 2048 * 2);   // 32 MB concat [tokens|att] bf16
  float*  convbuf = (float*) take((size_t)256 * 16 * 64 * 64 * 4); // 64 MB
  float*  qb      = (float*) take((size_t)8192 * 1024 * 4);   // 32 MB
  float*  kb      = (float*) take((size_t)8192 * 1024 * 4);
  float*  vb      = (float*) take((size_t)8192 * 1024 * 4);
  __bf16* ctxb    = (__bf16*)take((size_t)8192 * 1024 * 2);   // 16 MB
  float*  h1buf   = (float*) take((size_t)32 * 256 * 512 * 4);// 16 MB
  float*  eob     = (float*) take((size_t)256 * 2048 * 4);    // 2 MB
  __bf16* wqt     = (__bf16*)take((size_t)1024 * 1024 * 2);
  __bf16* wkt     = (__bf16*)take((size_t)1024 * 1024 * 2);
  __bf16* wvt     = (__bf16*)take((size_t)1024 * 1024 * 2);
  __bf16* wot     = (__bf16*)take((size_t)1024 * 1024 * 2);
  __bf16* W1t     = (__bf16*)take((size_t)32 * 512 * 2048 * 2); // 64 MB
  (void)in_sizes; (void)n_in; (void)out_size; (void)ws_size;

  // ---- weight transpose + bf16 convert ----
  {
    dim3 blk(32, 8);
    k_transpose_bf16<<<dim3(32, 32, 1), blk, 0, stream>>>(wq, 0, wqt, 0, 1024, 1024);
    k_transpose_bf16<<<dim3(32, 32, 1), blk, 0, stream>>>(wk, 0, wkt, 0, 1024, 1024);
    k_transpose_bf16<<<dim3(32, 32, 1), blk, 0, stream>>>(wv, 0, wvt, 0, 1024, 1024);
    k_transpose_bf16<<<dim3(32, 32, 1), blk, 0, stream>>>(wo, 0, wot, 0, 1024, 1024);
    k_transpose_bf16<<<dim3(512 / 32, 2048 / 32, 32), blk, 0, stream>>>(
        W1, (long)2048 * 512, W1t, (long)512 * 2048, 2048, 512);
  }

  // ---- CNN ----
  k_conv1<<<65536, 256, 0, stream>>>(x, c1w, c1b, convbuf);
  k_conv2<<<32768, 256, 16 * 18 * 66 * 4, stream>>>(convbuf, c2w, c2b, z);

  // ---- helper for GEMM launches ----
  auto gemm = [&](const __bf16* A, long lda, long ab,
                  const __bf16* Bt, long bb,
                  void* C, long ldc, long cb, int cbf16,
                  const float* bias, long biasb,
                  int M, int N, int K, int relu, int batch) {
    const int waves = (M / 32) * (N / 64);
    k_gemm_bf16<<<dim3(waves / 4, batch), 128, 0, stream>>>(
        A, lda, ab, Bt, (long)K, bb, C, ldc, cb, cbf16, bias, biasb, M, N, K, relu);
  };

  // ---- QKV projections: A = tokens (z[:, :1024], lda=2048) ----
  gemm(z, 2048, 0, wqt, 0, qb, 1024, 0, 0, nullptr, 0, 8192, 1024, 1024, 0, 1);
  gemm(z, 2048, 0, wkt, 0, kb, 1024, 0, 0, nullptr, 0, 8192, 1024, 1024, 0, 1);
  gemm(z, 2048, 0, wvt, 0, vb, 1024, 0, 0, nullptr, 0, 8192, 1024, 1024, 0, 1);

  // ---- attention -> ctx (bf16) ----
  k_attention<<<1024, 256, (8192 + 8192 + 1024) * 4, stream>>>(qb, kb, vb, ctxb);

  // ---- WO projection: writes bf16 into z[:, 1024:2048] ----
  gemm(ctxb, 1024, 0, wot, 0, z + 1024, 2048, 0, 1, nullptr, 0, 8192, 1024, 1024, 0, 1);

  // ---- expert W1 grouped GEMM: per expert e, A = z + e*2048 (lda = 32*2048) ----
  gemm(z, (long)32 * 2048, 2048,
       W1t, (long)512 * 2048,
       h1buf, 512, (long)256 * 512, 0,
       b1, 512,
       256, 512, 2048, 1, 32);

  // ---- expert W2 + final head ----
  k_expert2<<<2048, 256, 0, stream>>>(h1buf, W2, b2, eob);
  k_final<<<2, 256, 0, stream>>>(eob, Wf, bfv, out);
}
